// AsynBottleneck_13511967113615
// MI455X (gfx1250) — matrix operand order repaired and checked by reading, not compiled, not run on hardware
//
#include <hip/hip_runtime.h>

#define HGRID 768
#define WGRID 768
#define NSITES 200000
#define CIN 256
#define CB 64

typedef __bf16 bf16_t;
typedef __attribute__((ext_vector_type(16))) __bf16 v16bf;
typedef __attribute__((ext_vector_type(8)))  __bf16 v8bf;
typedef __attribute__((ext_vector_type(8)))  float  v8f;
typedef __attribute__((ext_vector_type(4)))  float  v4f;

__device__ __forceinline__ v16bf frag_zero() {
  v16bf r;
#pragma unroll
  for (int e = 0; e < 16; ++e) r[e] = (bf16_t)0.0f;
  return r;
}

__device__ __forceinline__ v16bf frag_join(v8bf lo, v8bf hi) {
  v16bf r;
#pragma unroll
  for (int e = 0; e < 8; ++e) { r[e] = lo[e]; r[8 + e] = hi[e]; }
  return r;
}

// A fragment: 16x32 bf16 tile, row-major source with the lane's row at `rowbase`.
// ISA 7.12.2: lanes 0-15 hold K = {k0..k0+7, k0+16..k0+23}, lanes 16-31 are +8.
__device__ __forceinline__ v16bf load_fragA(const bf16_t* rowbase, int k0, int lane) {
  const int hi = (lane >> 4) << 3;
  v8bf a = *(const v8bf*)(rowbase + k0 + hi);
  v8bf b = *(const v8bf*)(rowbase + k0 + 16 + hi);
  return frag_join(a, b);
}

// B fragment: 32x16 bf16 tile from transposed weights W^T[N][K] (kpitch = K dim).
// Lane holds column n0+(lane&15); K = {k0..k0+7, k0+16..k0+23}, lanes 16-31 are +8.
__device__ __forceinline__ v16bf load_fragB(const bf16_t* wt, int kpitch, int n0, int k0, int lane) {
  const bf16_t* p = wt + (size_t)(n0 + (lane & 15)) * kpitch + k0 + ((lane >> 4) << 3);
  v8bf a = *(const v8bf*)(p);
  v8bf b = *(const v8bf*)(p + 16);
  return frag_join(a, b);
}

__global__ void k_fill_idx(int* __restrict__ idx_map, int hw) {
  int i = blockIdx.x * blockDim.x + threadIdx.x;
  if (i < hw) idx_map[i] = -1;
}

__global__ void k_scatter(const int* __restrict__ coords, int* __restrict__ idx_map, int n) {
  int i = blockIdx.x * blockDim.x + threadIdx.x;
  if (i < n) idx_map[coords[i]] = i;
}

// Convert weights to bf16 and transpose to [N][K] layout for contiguous B-fragment loads.
__global__ void k_convert_weights(const float* __restrict__ w1, const float* __restrict__ w2,
                                  const float* __restrict__ w3,
                                  bf16_t* __restrict__ w1t, bf16_t* __restrict__ w2t,
                                  bf16_t* __restrict__ w3t) {
  int t = blockIdx.x * blockDim.x + threadIdx.x;
  if (t < CIN * CB) {                       // w1: [K=256][N=64] -> w1t[N][K]
    int k = t / CB, n = t - k * CB;
    w1t[n * CIN + k] = (bf16_t)w1[t];
  } else if (t < CIN * CB + 9 * CB * CB) {  // w2: [tap][K=64][N=64] -> w2t[tap][N][K]
    int j = t - CIN * CB;
    int tap = j / (CB * CB);
    int r = j - tap * (CB * CB);
    int kin = r / CB, nout = r - kin * CB;
    w2t[tap * CB * CB + nout * CB + kin] = (bf16_t)w2[j];
  } else if (t < CIN * CB + 9 * CB * CB + CB * CIN) {  // w3: [K=64][N=256] -> w3t[N][K]
    int j = t - CIN * CB - 9 * CB * CB;
    int k = j / CIN, n = j - k * CIN;
    w3t[n * CB + k] = (bf16_t)w3[j];
  }
}

// Stage 1: Y1 = relu((feats @ w1) * s1 + b1), bf16 output. 16 rows x 64 cols per block.
__global__ __launch_bounds__(128) void k_stage1(const float* __restrict__ feats,
                                                const bf16_t* __restrict__ w1t,
                                                const float* __restrict__ s1,
                                                const float* __restrict__ b1,
                                                bf16_t* __restrict__ y1) {
  __shared__ bf16_t ldsA[16 * CIN];  // 8 KB: 16x256 bf16 activation tile
  const int r0 = blockIdx.x * 16;
  const int tid = threadIdx.x;
  {
    // Cooperative streaming (non-temporal) load of 16x256 f32, convert to bf16 in LDS.
    const v4f* src = (const v4f*)(feats + (size_t)r0 * CIN);
#pragma unroll
    for (int j = 0; j < 8; ++j) {
      int q = tid + j * 128;  // float4 index in tile (1024 total)
      v4f v = __builtin_nontemporal_load(&src[q]);
      bf16_t* dst = &ldsA[q * 4];
#pragma unroll
      for (int e = 0; e < 4; ++e) dst[e] = (bf16_t)v[e];
    }
  }
  __syncthreads();
  const int wave = tid >> 5, lane = tid & 31;
  const int n0 = wave * 16;
  const bf16_t* rowbase = &ldsA[(lane & 15) * CIN];
  v8f acc = {};
#pragma unroll
  for (int k0 = 0; k0 < CIN; k0 += 32) {
    v16bf a = load_fragA(rowbase, k0, lane);
    v16bf b = load_fragB(w1t, CIN, n0, k0, lane);
    acc = __builtin_amdgcn_wmma_f32_16x16x32_bf16(false, a, false, b, (short)0, acc, false, false);
  }
  const int col = n0 + (lane & 15);
  const float sc = s1[col], bi = b1[col];
  const int rb = r0 + ((lane >> 4) << 3);
#pragma unroll
  for (int v = 0; v < 8; ++v) {
    float x = acc[v] * sc + bi;
    x = x > 0.f ? x : 0.f;
    y1[(size_t)(rb + v) * CB + col] = (bf16_t)x;
  }
}

// Stage 2: sparse 3x3 conv: Y2 = relu((sum_tap gather(Y1) @ w2[tap]) * s2 + b2).
// 32 rows x 64 cols per block; 8 waves = 2 row-tiles x 4 col-tiles. Y1 is L2-resident.
__global__ __launch_bounds__(256) void k_stage2(const bf16_t* __restrict__ y1,
                                                const int* __restrict__ coords,
                                                const int* __restrict__ idx_map,
                                                const bf16_t* __restrict__ w2t,
                                                const float* __restrict__ s2,
                                                const float* __restrict__ b2,
                                                bf16_t* __restrict__ y2) {
  __shared__ int nbr[32 * 9];
  const int r0g = blockIdx.x * 32;
  const int tid = threadIdx.x;
  for (int e = tid; e < 32 * 9; e += 256) {
    const int i = e / 9, tap = e - i * 9;
    const int c = coords[r0g + i];
    const int y = c / WGRID, x = c - y * WGRID;
    const int ny = y + tap / 3 - 1, nx = x + tap % 3 - 1;
    int r = -1;
    if (ny >= 0 && ny < HGRID && nx >= 0 && nx < WGRID) r = idx_map[ny * WGRID + nx];
    nbr[e] = r;
  }
  __syncthreads();
  const int wave = tid >> 5, lane = tid & 31;
  const int rt = wave >> 2, ct = wave & 3;
  const int n0 = ct * 16;
  v8f acc = {};
#pragma unroll
  for (int tap = 0; tap < 9; ++tap) {
    const int ridx = nbr[((rt << 4) + (lane & 15)) * 9 + tap];
    const bf16_t* wt = w2t + tap * CB * CB;
#pragma unroll
    for (int k0 = 0; k0 < CB; k0 += 32) {
      v16bf a = (ridx >= 0) ? load_fragA(y1 + (size_t)ridx * CB, k0, lane) : frag_zero();
      v16bf b = load_fragB(wt, CB, n0, k0, lane);
      acc = __builtin_amdgcn_wmma_f32_16x16x32_bf16(false, a, false, b, (short)0, acc, false, false);
    }
  }
  const int col = n0 + (lane & 15);
  const float sc = s2[col], bi = b2[col];
  const int rb = r0g + (rt << 4) + ((lane >> 4) << 3);
#pragma unroll
  for (int v = 0; v < 8; ++v) {
    float x = acc[v] * sc + bi;
    x = x > 0.f ? x : 0.f;
    y2[(size_t)(rb + v) * CB + col] = (bf16_t)x;
  }
}

// Stage 3: out = relu((Y2 @ w3) * s3 + b3 + feats). 16 rows x 128 cols per block (8 waves).
__global__ __launch_bounds__(256) void k_stage3(const bf16_t* __restrict__ y2,
                                                const bf16_t* __restrict__ w3t,
                                                const float* __restrict__ s3,
                                                const float* __restrict__ b3,
                                                const float* __restrict__ feats,
                                                float* __restrict__ out) {
  const int r0 = blockIdx.x * 16;
  const int tid = threadIdx.x;
  const int wave = tid >> 5, lane = tid & 31;
  const int n0 = (blockIdx.y * 8 + wave) * 16;
  const bf16_t* rowbase = y2 + (size_t)(r0 + (lane & 15)) * CB;
  v8f acc = {};
#pragma unroll
  for (int k0 = 0; k0 < CB; k0 += 32) {
    v16bf a = load_fragA(rowbase, k0, lane);
    v16bf b = load_fragB(w3t, CB, n0, k0, lane);
    acc = __builtin_amdgcn_wmma_f32_16x16x32_bf16(false, a, false, b, (short)0, acc, false, false);
  }
  const int col = n0 + (lane & 15);
  const float sc = s3[col], bi = b3[col];
  const int rb = r0 + ((lane >> 4) << 3);
#pragma unroll
  for (int v = 0; v < 8; ++v) {
    // Residual read + output write are streamed once -> non-temporal to preserve L2.
    float res = __builtin_nontemporal_load(&feats[(size_t)(rb + v) * CIN + col]);
    float x = acc[v] * sc + bi + res;
    x = x > 0.f ? x : 0.f;
    __builtin_nontemporal_store(x, &out[(size_t)(rb + v) * CIN + col]);
  }
}

extern "C" void kernel_launch(void* const* d_in, const int* in_sizes, int n_in,
                              void* d_out, int out_size, void* d_ws, size_t ws_size,
                              hipStream_t stream) {
  (void)in_sizes; (void)n_in; (void)out_size; (void)ws_size;
  const float* feats = (const float*)d_in[0];
  const int*   coords = (const int*)d_in[1];
  const float* w1 = (const float*)d_in[2];
  const float* w2 = (const float*)d_in[3];
  const float* w3 = (const float*)d_in[4];
  const float* s1 = (const float*)d_in[5];
  const float* b1 = (const float*)d_in[6];
  const float* s2 = (const float*)d_in[7];
  const float* b2 = (const float*)d_in[8];
  const float* s3 = (const float*)d_in[9];
  const float* b3 = (const float*)d_in[10];
  float* out = (float*)d_out;

  char* ws = (char*)d_ws;
  size_t off = 0;
  auto take = [&](size_t bytes) -> char* {
    char* p = ws + off;
    off = (off + bytes + 255) & ~(size_t)255;
    return p;
  };
  int*    idx_map = (int*)take((size_t)HGRID * WGRID * sizeof(int));      // 2.36 MB
  bf16_t* y1      = (bf16_t*)take((size_t)NSITES * CB * sizeof(bf16_t));  // 25.6 MB
  bf16_t* y2      = (bf16_t*)take((size_t)NSITES * CB * sizeof(bf16_t));  // 25.6 MB
  bf16_t* w1t     = (bf16_t*)take((size_t)CIN * CB * sizeof(bf16_t));
  bf16_t* w2t     = (bf16_t*)take((size_t)9 * CB * CB * sizeof(bf16_t));
  bf16_t* w3t     = (bf16_t*)take((size_t)CB * CIN * sizeof(bf16_t));

  const int hw = HGRID * WGRID;
  hipLaunchKernelGGL(k_fill_idx, dim3((hw + 255) / 256), dim3(256), 0, stream, idx_map, hw);
  hipLaunchKernelGGL(k_scatter, dim3((NSITES + 255) / 256), dim3(256), 0, stream,
                     coords, idx_map, NSITES);
  const int wtot = CIN * CB + 9 * CB * CB + CB * CIN;
  hipLaunchKernelGGL(k_convert_weights, dim3((wtot + 255) / 256), dim3(256), 0, stream,
                     w1, w2, w3, w1t, w2t, w3t);
  hipLaunchKernelGGL(k_stage1, dim3(NSITES / 16), dim3(128), 0, stream, feats, w1t, s1, b1, y1);
  hipLaunchKernelGGL(k_stage2, dim3(NSITES / 32), dim3(256), 0, stream,
                     y1, coords, idx_map, w2t, s2, b2, y2);
  hipLaunchKernelGGL(k_stage3, dim3(NSITES / 16, 2), dim3(256), 0, stream,
                     y2, w3t, s3, b3, feats, out);
}
